// GCN_36155034698159
// MI455X (gfx1250) — matrix operand order repaired and checked by reading, not compiled, not run on hardware
//
#include <hip/hip_runtime.h>
#include <hip/hip_bf16.h>
#include <stdint.h>

// Problem sizes (fixed by the reference)
#define NN 50000      // nodes
#define FE 512        // features == hidden
#define HD 512
#define NC 64         // classes
#define NE 800000     // edges
#define MPAD 50048    // 391 * 128, row padding for the 128-row GEMM blocks

typedef __attribute__((ext_vector_type(16))) __bf16 v16bf;
typedef __attribute__((ext_vector_type(8)))  __bf16 v8bf;
typedef __attribute__((ext_vector_type(8)))  float  v8f;

// ---------------- workspace layout (bytes) ----------------
static constexpr size_t SZ_DEG  = (size_t)MPAD * 4;            // f32 deg -> dis (in place)
static constexpr size_t SZ_ACT  = (size_t)MPAD * HD * 2;       // bf16 activations
static constexpr size_t SZ_H    = (size_t)MPAD * HD * 4;       // f32 GEMM output
static constexpr size_t SZ_W    = (size_t)HD * HD * 2;         // bf16 512x512 transposed weight
static constexpr size_t SZ_LW2  = (size_t)NC * HD * 2;         // bf16 64x512 transposed weight

static constexpr size_t OFF_DEG  = 0;
static constexpr size_t OFF_ACT  = OFF_DEG  + SZ_DEG;
static constexpr size_t OFF_H    = OFF_ACT  + SZ_ACT;
static constexpr size_t OFF_AGG  = OFF_H    + SZ_H;
static constexpr size_t OFF_W1T  = OFF_AGG  + SZ_H;
static constexpr size_t OFF_W2T  = OFF_W1T  + SZ_W;
static constexpr size_t OFF_W3T  = OFF_W2T  + SZ_W;
static constexpr size_t OFF_LW1T = OFF_W3T  + SZ_W;
static constexpr size_t OFF_LW2T = OFF_LW1T + SZ_W;
static constexpr size_t WS_NEED  = OFF_LW2T + SZ_LW2;

// ---------------- degree / normalization ----------------
__global__ __launch_bounds__(256) void k_init_deg(float* __restrict__ deg) {
  int i = blockIdx.x * 256 + threadIdx.x;
  if (i < NN) deg[i] = 1.0f;   // self-loop contributes exactly 1
}

__global__ __launch_bounds__(256) void k_accum_deg(const long long* __restrict__ ei,
                                                   float* __restrict__ deg) {
  int e = blockIdx.x * 256 + threadIdx.x;
  if (e < NE) {
    int d = (int)ei[NE + e];   // dst row of edge_index
    __hip_atomic_fetch_add(&deg[d], 1.0f, __ATOMIC_RELAXED, __HIP_MEMORY_SCOPE_AGENT);
  }
}

__global__ __launch_bounds__(256) void k_deg_to_dis(float* __restrict__ deg) {
  int i = blockIdx.x * 256 + threadIdx.x;
  if (i < NN) deg[i] = rsqrtf(deg[i]);   // deg >= 1 always
}

// ---------------- fp32 -> bf16 conversions ----------------
__global__ __launch_bounds__(256) void k_cvt_x(const float* __restrict__ x,
                                               __bf16* __restrict__ act) {
  size_t i = (size_t)blockIdx.x * 256 + threadIdx.x;   // over MPAD*FE
  float v = (i < (size_t)NN * FE) ? x[i] : 0.0f;       // zero the pad rows once
  act[i] = (__bf16)v;
}

// W is [K x N] row-major fp32; WT is [N x K] bf16 (so B fragments load contiguously)
__global__ __launch_bounds__(256) void k_cvt_wT(const float* __restrict__ W,
                                                __bf16* __restrict__ WT, int K, int N) {
  int idx = blockIdx.x * 256 + threadIdx.x;
  if (idx >= K * N) return;
  int n = idx / K, k = idx % K;
  WT[idx] = (__bf16)W[(size_t)k * N + n];
}

// ---------------- WMMA bf16 GEMM: C[M_PAD x N] = A[M_PAD x K] * BT^T ----------------
// Block: 256 threads = 8 waves. Block tile: 128 rows x 64 cols. Wave tile: 16 x 64.
// The shared 64x32 B tile is staged through LDS with double-buffered
// GLOBAL_LOAD_ASYNC_TO_LDS_B128 (ASYNCcnt) so the 8 waves don't re-load it 8x from L2.
__global__ __launch_bounds__(256) void k_gemm_bf16(const __bf16* __restrict__ A,
                                                   const __bf16* __restrict__ BT, // [N x K]
                                                   float* __restrict__ C,
                                                   int K, int N) {
  __shared__ __align__(16) __bf16 sB[2][64][32];   // 2 x 4KB double buffer

  const int lane = threadIdx.x & 31;
  const int wave = threadIdx.x >> 5;
  const int hf   = lane >> 4;      // which 16-lane half
  const int l16  = lane & 15;

  const int row0 = blockIdx.y * 128 + wave * 16;
  const int col0 = blockIdx.x * 64;

  // async-copy mapping: 256 threads x 16B = one 64col x 32K bf16 tile per step
  const int tc = threadIdx.x >> 2;   // column 0..63
  const int tq = threadIdx.x & 3;    // 16B chunk 0..3 within the column's 64B
  const __bf16* __restrict__ gB = BT + (size_t)(col0 + tc) * K + tq * 8;

  v8f acc[4] = {};
  const __bf16* __restrict__ arow = A + (size_t)(row0 + l16) * K;

  // prefetch K-step 0 into buffer 0
  {
    uint64_t ga = (uint64_t)(uintptr_t)gB;
    uint32_t la = (uint32_t)(uintptr_t)&sB[0][tc][tq * 8];
    asm volatile("global_load_async_to_lds_b128 %0, %1, off"
                 :: "v"(la), "v"(ga) : "memory");
  }

  const int nstep = K >> 5;
  for (int s = 0; s < nstep; ++s) {
    const int buf = s & 1;
    if (s + 1 < nstep) {
      // prefetch next B tile into the other buffer, then wait for current one
      uint64_t ga = (uint64_t)(uintptr_t)(gB + ((s + 1) << 5));
      uint32_t la = (uint32_t)(uintptr_t)&sB[buf ^ 1][tc][tq * 8];
      asm volatile("global_load_async_to_lds_b128 %0, %1, off"
                   :: "v"(la), "v"(ga) : "memory");
      asm volatile("s_wait_asynccnt 0x1" ::: "memory");
    } else {
      asm volatile("s_wait_asynccnt 0x0" ::: "memory");
    }
    __syncthreads();   // current buffer populated by all waves

    const int k0 = s << 5;
    // A fragment (16x32): lane l16 = row; elems 0..15 = K in [k0+16*hf, +16)
    union { v16bf v; v8bf h[2]; } af;
    af.h[0] = *(const v8bf*)(arow + k0 + 16 * hf);
    af.h[1] = *(const v8bf*)(arow + k0 + 16 * hf + 8);

#pragma unroll
    for (int t = 0; t < 4; ++t) {
      // B fragment (32x16): lane l16 = col; 16 contiguous K values at 16*hf (32B LDS read)
      v16bf bf = *(const v16bf*)&sB[buf][t * 16 + l16][16 * hf];
      acc[t] = __builtin_amdgcn_wmma_f32_16x16x32_bf16(
          false, af.v, false, bf, (short)0, acc[t], false, false);
    }
    __syncthreads();   // everyone done reading buf before it is overwritten (s+2)
  }

#pragma unroll
  for (int t = 0; t < 4; ++t)
#pragma unroll
    for (int j = 0; j < 8; ++j) {
      int r = row0 + j + 8 * hf;          // C/D layout: VGPR j -> row j + 8*half
      int c = col0 + t * 16 + l16;
      C[(size_t)r * N + c] = acc[t][j];
    }
}

// ---------------- GCN aggregation ----------------
// Self-loop term initializes the accumulator: AGG[i,f] = dis[i]^2 * H[i,f]
__global__ __launch_bounds__(256) void k_agg_init(const float* __restrict__ dis,
                                                  const float* __restrict__ H,
                                                  float* __restrict__ AGG) {
  size_t idx = (size_t)blockIdx.x * 256 + threadIdx.x;  // over NN*HD
  int i = (int)(idx >> 9);
  float w = dis[i];
  AGG[idx] = w * w * H[idx];
}

// One wave per edge; lanes stride the 512 features. L2-resident gather + f32 atomics.
__global__ __launch_bounds__(256) void k_agg_edges(const long long* __restrict__ ei,
                                                   const float* __restrict__ dis,
                                                   const float* __restrict__ H,
                                                   float* __restrict__ AGG) {
  int e = blockIdx.x * 8 + (threadIdx.x >> 5);
  if (e >= NE) return;
  int lane = threadIdx.x & 31;
  long long s = ei[e];
  long long d = ei[NE + e];
  float w = dis[s] * dis[d];
  const float* __restrict__ hs = H + (size_t)s * HD;
  float* __restrict__ od = AGG + (size_t)d * HD;
#pragma unroll
  for (int f = lane; f < HD; f += 32)
    __hip_atomic_fetch_add(&od[f], w * hs[f], __ATOMIC_RELAXED, __HIP_MEMORY_SCOPE_AGENT);
}

// bias + ReLU, write bf16 activations for the next GEMM
__global__ __launch_bounds__(256) void k_bias_relu_bf(const float* __restrict__ IN,
                                                      const float* __restrict__ b,
                                                      __bf16* __restrict__ ACT) {
  size_t idx = (size_t)blockIdx.x * 256 + threadIdx.x;  // over NN*HD
  int f = (int)(idx & (HD - 1));
  float v = IN[idx] + b[f];
  ACT[idx] = (__bf16)(v > 0.0f ? v : 0.0f);
}

// ---------------- log-softmax over 64 classes, one wave per row ----------------
__global__ __launch_bounds__(256) void k_logsoftmax(const float* __restrict__ L,
                                                    const float* __restrict__ b,
                                                    float* __restrict__ out) {
  int row = blockIdx.x * 8 + (threadIdx.x >> 5);
  if (row >= NN) return;
  int lane = threadIdx.x & 31;
  const float* lr = L + (size_t)row * NC;
  float v0 = lr[lane] + b[lane];
  float v1 = lr[32 + lane] + b[32 + lane];
  float m = fmaxf(v0, v1);
#pragma unroll
  for (int o = 16; o; o >>= 1) m = fmaxf(m, __shfl_xor(m, o, 32));
  float s = expf(v0 - m) + expf(v1 - m);
#pragma unroll
  for (int o = 16; o; o >>= 1) s += __shfl_xor(s, o, 32);
  float ls = logf(s);
  float* orow = out + (size_t)row * NC;
  orow[lane]      = v0 - m - ls;
  orow[32 + lane] = v1 - m - ls;
}

// ---------------- driver ----------------
extern "C" void kernel_launch(void* const* d_in, const int* in_sizes, int n_in,
                              void* d_out, int out_size, void* d_ws, size_t ws_size,
                              hipStream_t stream) {
  const float*     x   = (const float*)d_in[0];
  const long long* ei  = (const long long*)d_in[1];
  const float*     W1  = (const float*)d_in[2];
  const float*     b1  = (const float*)d_in[3];
  const float*     W2  = (const float*)d_in[4];
  const float*     b2  = (const float*)d_in[5];
  const float*     W3  = (const float*)d_in[6];
  const float*     b3  = (const float*)d_in[7];
  const float*     lw1 = (const float*)d_in[8];
  const float*     lb1 = (const float*)d_in[9];
  const float*     lw2 = (const float*)d_in[10];
  const float*     lb2 = (const float*)d_in[11];
  float* out = (float*)d_out;

  if (ws_size < WS_NEED) return;  // need ~247 MiB of scratch

  char* ws = (char*)d_ws;
  float*  DIS  = (float*) (ws + OFF_DEG);
  __bf16* ACT  = (__bf16*)(ws + OFF_ACT);
  float*  H    = (float*) (ws + OFF_H);
  float*  AGG  = (float*) (ws + OFF_AGG);
  __bf16* W1T  = (__bf16*)(ws + OFF_W1T);
  __bf16* W2T  = (__bf16*)(ws + OFF_W2T);
  __bf16* W3T  = (__bf16*)(ws + OFF_W3T);
  __bf16* LW1T = (__bf16*)(ws + OFF_LW1T);
  __bf16* LW2T = (__bf16*)(ws + OFF_LW2T);

  const int ELT_NN  = (NN * HD) / 256;          // 100000 blocks (exact)
  const int ELT_PAD = (int)(((size_t)MPAD * HD) / 256);

  // normalization: deg (with self loop) -> dis = rsqrt(deg)
  k_init_deg  <<<(NN + 255) / 256, 256, 0, stream>>>(DIS);
  k_accum_deg <<<(NE + 255) / 256, 256, 0, stream>>>(ei, DIS);
  k_deg_to_dis<<<(NN + 255) / 256, 256, 0, stream>>>(DIS);

  // precision conversion
  k_cvt_x <<<ELT_PAD, 256, 0, stream>>>(x, ACT);
  k_cvt_wT<<<(HD * HD + 255) / 256, 256, 0, stream>>>(W1,  W1T,  HD, HD);
  k_cvt_wT<<<(HD * HD + 255) / 256, 256, 0, stream>>>(W2,  W2T,  HD, HD);
  k_cvt_wT<<<(HD * HD + 255) / 256, 256, 0, stream>>>(W3,  W3T,  HD, HD);
  k_cvt_wT<<<(HD * HD + 255) / 256, 256, 0, stream>>>(lw1, LW1T, HD, HD);
  k_cvt_wT<<<(HD * NC + 255) / 256, 256, 0, stream>>>(lw2, LW2T, HD, NC);

  const dim3 gemmGrid(HD / 64, MPAD / 128);   // (8, 391)
  const __bf16* WT[3]   = {W1T, W2T, W3T};
  const float*  bias[3] = {b1, b2, b3};

  // 3 GCN conv layers
  for (int l = 0; l < 3; ++l) {
    k_gemm_bf16 <<<gemmGrid, 256, 0, stream>>>(ACT, WT[l], H, HD, HD);
    k_agg_init  <<<ELT_NN, 256, 0, stream>>>(DIS, H, AGG);
    k_agg_edges <<<NE / 8, 256, 0, stream>>>(ei, DIS, H, AGG);
    k_bias_relu_bf<<<ELT_NN, 256, 0, stream>>>(AGG, bias[l], ACT);
  }

  // MLP head
  k_gemm_bf16  <<<gemmGrid, 256, 0, stream>>>(ACT, LW1T, H, HD, HD);
  k_bias_relu_bf<<<ELT_NN, 256, 0, stream>>>(H, lb1, ACT);
  k_gemm_bf16  <<<dim3(NC / 64, MPAD / 128), 256, 0, stream>>>(ACT, LW2T, H, HD, NC);
  k_logsoftmax <<<(NN + 7) / 8, 256, 0, stream>>>(H, lb2, out);
}
